// DNeRFRenderer_2594160246826
// MI455X (gfx1250) — matrix operand order repaired and checked by reading, compiled, not run on hardware
//
#include <hip/hip_runtime.h>
#include <hip/hip_bf16.h>
#include <math.h>

// ---------------- constants (match reference) ----------------
#define N_RAYS 8192
#define S_COARSE 64
#define S_FINE 64
#define S_ALL 128
#define HID 64
#define GEO 15
#define MIN_NEAR 0.05f

typedef _Float16 v16h __attribute__((ext_vector_type(16)));
typedef _Float16 v8h  __attribute__((ext_vector_type(8)));
typedef float    v8f  __attribute__((ext_vector_type(8)));

// Number of precomputed B fragments: density L1 (4 N-tiles), density L2 (2 K-chunks),
// color L1 (4 N-tiles), color L2 (2 K-chunks).
#define NFRAG 12
#define FRAG_HALVES (32 * 16)  // per fragment: 32 lanes x 16 halves

// ---------------- WMMA helpers ----------------
static __device__ inline v8f wmma_f32_16x16x32_f16(v16h a, v16h b, v8f c) {
  // (neg_a, A, neg_b, B, c_mod, C, reuse_a, reuse_b)
  return __builtin_amdgcn_wmma_f32_16x16x32_f16(false, a, false, b, (short)0, c,
                                                false, false);
}

static __device__ inline v8f v8f_zero() {
  v8f c;
#pragma unroll
  for (int i = 0; i < 8; ++i) c[i] = 0.0f;
  return c;
}

static __device__ inline v16h join16(v8h lo, v8h hi) {
  v16h r;
#pragma unroll
  for (int i = 0; i < 8; ++i) { r[i] = lo[i]; r[8 + i] = hi[i]; }
  return r;
}

// Build a 32x16 f16 B-fragment from a row-major fp32 weight matrix W[krows x ldn].
// Rows >= krows and cols >= ncols are zero-padded.
// Layout: lane&15 = N column; lanes 0-15 carry K=k0..k0+15, lanes 16-31 K=k0+16..k0+31;
// half h of VGPR v = K = base + 2v + h.
static __device__ inline v16h bfrag_from_f32(const float* __restrict__ W, int ldn,
                                             int krows, int ncols, int n0, int k0,
                                             int lane) {
  const int n  = n0 + (lane & 15);
  const int kb = k0 + ((lane >> 4) << 4);
  v16h b;
#pragma unroll
  for (int v = 0; v < 8; ++v) {
    const int ka = kb + 2 * v;
    const int kc = ka + 1;
    float f0 = (ka < krows && n < ncols) ? W[ka * ldn + n] : 0.0f;
    float f1 = (kc < krows && n < ncols) ? W[kc * ldn + n] : 0.0f;
    b[2 * v]     = (_Float16)f0;
    b[2 * v + 1] = (_Float16)f1;
  }
  return b;
}

// Load one precomputed lane-major B fragment: two coalesced 16B loads.
static __device__ inline v16h load_frag(const _Float16* __restrict__ fragbuf,
                                        int f, int lane) {
  const _Float16* p = fragbuf + ((size_t)f * 32 + lane) * 16;
  v8h lo = *reinterpret_cast<const v8h*>(p);
  v8h hi = *reinterpret_cast<const v8h*>(p + 8);
  return join16(lo, hi);
}

// Load a 16x32 f16 A-fragment from a half buffer with rowStride halves per row.
// Lane&15 = M row. lanes 0-15: halves = K k0+0..15; lanes 16-31:
// K k0+16..31.  Offsets are 8-half (16B) aligned.
static __device__ inline v16h afrag_from_half(const _Float16* __restrict__ buf,
                                              int rowStride, int row, int k0,
                                              int lane) {
  const int sel  = lane >> 4;
  const int base = row * rowStride + k0 + sel * 16;
  v8h lo = *reinterpret_cast<const v8h*>(buf + base);
  v8h hi = *reinterpret_cast<const v8h*>(buf + base + 8);
  return join16(lo, hi);
}

// ================= K0: pack all weight B-fragments (once per launch) =========
__global__ void k_pack(const float* __restrict__ Wd1, const float* __restrict__ Wd2,
                       const float* __restrict__ Wc1, const float* __restrict__ Wc2,
                       _Float16* __restrict__ fragbuf) {
  const int f = blockIdx.x;       // 12 blocks
  const int lane = threadIdx.x;   // 32 threads = 1 wave
  const float* W;
  int ldn, krows, ncols, n0, k0;
  if (f < 4)       { W = Wd1; ldn = HID; krows = 4;  ncols = HID; n0 = f * 16;       k0 = 0; }
  else if (f < 6)  { W = Wd2; ldn = 16;  krows = HID; ncols = 16; n0 = 0;            k0 = (f - 4) * 32; }
  else if (f < 10) { W = Wc1; ldn = HID; krows = 3 + GEO; ncols = HID; n0 = (f - 6) * 16; k0 = 0; }
  else             { W = Wc2; ldn = 3;   krows = HID; ncols = 3;  n0 = 0;            k0 = (f - 10) * 32; }
  v16h b = bfrag_from_f32(W, ldn, krows, ncols, n0, k0, lane);
  _Float16* dst = fragbuf + ((size_t)f * 32 + lane) * 16;
  v8h lo, hi;
#pragma unroll
  for (int i = 0; i < 8; ++i) { lo[i] = b[i]; hi[i] = b[8 + i]; }
  *reinterpret_cast<v8h*>(dst)     = lo;
  *reinterpret_cast<v8h*>(dst + 8) = hi;
}

// ================= K1: near/far + coarse z =================
__global__ void k_prep(const float* __restrict__ rays_o,
                       const float* __restrict__ rays_d,
                       float* __restrict__ nears, float* __restrict__ fars,
                       float* __restrict__ z_coarse) {
  const int r = blockIdx.x * blockDim.x + threadIdx.x;
  if (r >= N_RAYS) return;
  float tmin = -3.0e38f, tmax = 3.0e38f;
#pragma unroll
  for (int a = 0; a < 3; ++a) {
    float o = rays_o[r * 3 + a];
    float d = rays_d[r * 3 + a];
    float dd = (fabsf(d) < 1e-9f) ? 1e-9f : d;
    float inv = 1.0f / dd;
    float t0 = (-1.0f - o) * inv;
    float t1 = ( 1.0f - o) * inv;
    tmin = fmaxf(tmin, fminf(t0, t1));
    tmax = fminf(tmax, fmaxf(t0, t1));
  }
  float nr = fmaxf(tmin, MIN_NEAR);
  float fr = (tmax < nr) ? (nr + 1e-2f) : tmax;
  nears[r] = nr;
  fars[r]  = fr;
  const float span = fr - nr;
#pragma unroll 4
  for (int s = 0; s < S_COARSE; ++s)
    z_coarse[r * S_COARSE + s] = nr + span * ((float)s * (1.0f / 63.0f));
}

// ================= K2/K4: density MLP (4 -> 64 -> 1+15) via WMMA =================
__global__ __launch_bounds__(256) void k_density(
    const float* __restrict__ rays_o, const float* __restrict__ rays_d,
    const float* __restrict__ timep,
    const _Float16* __restrict__ fragbuf,
    const float* __restrict__ bd1, const float* __restrict__ bd2,
    const float* __restrict__ zvals,       // [N_RAYS*64]
    float* __restrict__ sigma_out,         // [N_RAYS*64]
    _Float16* __restrict__ geo_out) {      // [N_RAYS*64*16] (col0 unused)
  __shared__ __align__(16) _Float16 ldsH[8 * 16 * HID];
  const int lane = threadIdx.x & 31;
  const int wave = threadIdx.x >> 5;
  const int tile = blockIdx.x * 8 + wave;
  const int m    = lane & 15;
  const int sel  = lane >> 4;

  // ---- A fragment (branch-free): rows = samples, K = [x,y,z,t,0,...,0] ----
  const int sample = tile * 16 + m;
  const int ray = sample >> 6;
  const float z = zvals[sample];
  float x = rays_o[ray * 3 + 0] + rays_d[ray * 3 + 0] * z;
  float y = rays_o[ray * 3 + 1] + rays_d[ray * 3 + 1] * z;
  float w = rays_o[ray * 3 + 2] + rays_d[ray * 3 + 2] * z;
  x = fminf(fmaxf(x, -1.0f), 1.0f);
  y = fminf(fmaxf(y, -1.0f), 1.0f);
  w = fminf(fmaxf(w, -1.0f), 1.0f);
  const float tt = timep[0];
  v16h a;
#pragma unroll
  for (int i = 0; i < 16; ++i) a[i] = (_Float16)0.0f;
  a[0] = sel ? (_Float16)0.0f : (_Float16)x;
  a[1] = sel ? (_Float16)0.0f : (_Float16)y;
  a[2] = sel ? (_Float16)0.0f : (_Float16)w;
  a[3] = sel ? (_Float16)0.0f : (_Float16)tt;

  // ---- Layer 1: 4x v_wmma_f32_16x16x32_f16 ----
  _Float16* myLds = ldsH + wave * 16 * HID;
#pragma unroll
  for (int nt = 0; nt < 4; ++nt) {
    v16h b = load_frag(fragbuf, nt, lane);
    v8f h = wmma_f32_16x16x32_f16(a, b, v8f_zero());
    const int n = nt * 16 + m;
    const float bb = bd1[n];
#pragma unroll
    for (int i = 0; i < 8; ++i) {
      float v = fmaxf(h[i] + bb, 0.0f);             // ReLU
      myLds[(sel * 8 + i) * HID + n] = (_Float16)v; // stage for transpose
    }
  }
  __syncthreads();

  // ---- Layer 2: K=64 in two chunks ----
  v8f o = v8f_zero();
#pragma unroll
  for (int c = 0; c < 2; ++c) {
    v16h a2 = afrag_from_half(myLds, HID, m, c * 32, lane);
    v16h b2 = load_frag(fragbuf, 4 + c, lane);
    o = wmma_f32_16x16x32_f16(a2, b2, o);
  }
  const int col = m;
  const float bo = bd2[col];
#pragma unroll
  for (int i = 0; i < 8; ++i) {
    const int row = sel * 8 + i;
    const int smp = tile * 16 + row;
    const float v = o[i] + bo;
    if (col == 0) sigma_out[smp] = expf(v);
    else          geo_out[(size_t)smp * 16 + col] = (_Float16)v;
  }
}

// ================= K3: importance sampling (per-ray scalar) =================
__global__ void k_importance(const float* __restrict__ z_coarse,
                             const float* __restrict__ sigma_coarse,
                             const float* __restrict__ nears,
                             const float* __restrict__ fars,
                             float* __restrict__ z_fine) {
  const int r = blockIdx.x * blockDim.x + threadIdx.x;
  if (r >= N_RAYS) return;
  const float sd = (fars[r] - nears[r]) * (1.0f / (float)S_COARSE);

  float z[S_COARSE];
  for (int s = 0; s < S_COARSE; ++s) z[s] = z_coarse[r * S_COARSE + s];

  float zm[63];
  float ww[62];
  float T = 1.0f;
  for (int s = 0; s < S_COARSE; ++s) {
    const float delta = (s < 63) ? (z[s + 1] - z[s]) : sd;
    const float sig = sigma_coarse[r * S_COARSE + s];
    const float alpha = 1.0f - expf(-delta * sig);
    const float w = alpha * T;
    T *= (1.0f - alpha + 1e-15f);
    if (s < 63) zm[s] = z[s] + 0.5f * delta;
    if (s >= 1 && s <= 62) ww[s - 1] = w + 1e-5f;
  }
  float sum = 0.0f;
  for (int j = 0; j < 62; ++j) sum += ww[j];
  const float inv = 1.0f / sum;
  float cdf[63];
  cdf[0] = 0.0f;
  for (int j = 0; j < 62; ++j) cdf[j + 1] = cdf[j] + ww[j] * inv;

  const float u0 = 0.5f / (float)S_FINE;
  const float du = (1.0f - 1.0f / (float)S_FINE) / 63.0f;
  for (int jj = 0; jj < S_FINE; ++jj) {
    const float u = u0 + du * (float)jj;
    int idx = 0;
    for (int q = 0; q < 63; ++q) idx += (cdf[q] <= u) ? 1 : 0;  // searchsorted right
    int below = idx - 1; below = below < 0 ? 0 : (below > 62 ? 62 : below);
    int above = idx;     above = above > 62 ? 62 : above;
    const float c0 = cdf[below], c1 = cdf[above];
    const float b0 = zm[below],  b1 = zm[above];
    float den = c1 - c0;
    den = (den < 1e-5f) ? 1.0f : den;
    const float t = (u - c0) / den;
    z_fine[r * S_FINE + jj] = b0 + t * (b1 - b0);
  }
}

// ================= K5: merge + weights + pack color-MLP inputs =================
__global__ void k_merge(const float* __restrict__ z_coarse,
                        const float* __restrict__ z_fine,
                        const float* __restrict__ sigma_coarse,
                        const float* __restrict__ sigma_fine,
                        const _Float16* __restrict__ geo_coarse,
                        const _Float16* __restrict__ geo_fine,
                        const float* __restrict__ rays_d,
                        const float* __restrict__ nears,
                        const float* __restrict__ fars,
                        _Float16* __restrict__ colin,   // [N*128*32]
                        float* __restrict__ wts) {      // [N*128]
  const int r = blockIdx.x * blockDim.x + threadIdx.x;
  if (r >= N_RAYS) return;
  const float sd = (fars[r] - nears[r]) * (1.0f / (float)S_COARSE);

  float za[S_COARSE], zb[S_FINE];
  for (int s = 0; s < S_COARSE; ++s) za[s] = z_coarse[r * S_COARSE + s];
  for (int s = 0; s < S_FINE; ++s)   zb[s] = z_fine[r * S_FINE + s];

  float zall[S_ALL];
  int   take[S_ALL];  // src*64 + idx
  {
    int i = 0, j = 0;
    for (int k = 0; k < S_ALL; ++k) {
      bool fromA = (j >= S_FINE) || (i < S_COARSE && za[i] <= zb[j]);
      if (fromA) { zall[k] = za[i]; take[k] = i;      ++i; }
      else       { zall[k] = zb[j]; take[k] = 64 + j; ++j; }
    }
  }

  const _Float16 hdx = (_Float16)rays_d[r * 3 + 0];
  const _Float16 hdy = (_Float16)rays_d[r * 3 + 1];
  const _Float16 hdz = (_Float16)rays_d[r * 3 + 2];
  const _Float16 h0  = (_Float16)0.0f;

  float T = 1.0f;
  for (int k = 0; k < S_ALL; ++k) {
    const float delta = (k < S_ALL - 1) ? (zall[k + 1] - zall[k]) : sd;
    const int tk = take[k];
    const int src = tk >> 6;
    const int idx = tk & 63;
    const float sig = src ? sigma_fine[r * S_FINE + idx]
                          : sigma_coarse[r * S_COARSE + idx];
    const float alpha = 1.0f - expf(-delta * sig);
    const float w = alpha * T;
    T *= (1.0f - alpha + 1e-15f);
    wts[(size_t)r * S_ALL + k] = w;

    // gather geo (two aligned 16B loads) and pack the 32-half color-MLP row
    const _Float16* g = (src ? geo_fine : geo_coarse) + ((size_t)r * 64 + idx) * 16;
    v8h g0 = *reinterpret_cast<const v8h*>(g);       // halves 0..7  (g[0] unused)
    v8h g1 = *reinterpret_cast<const v8h*>(g + 8);   // halves 8..15
    v8h o0, o1, o2, o3;
    o0[0] = hdx;  o0[1] = hdy;  o0[2] = hdz;
    o0[3] = g0[1]; o0[4] = g0[2]; o0[5] = g0[3]; o0[6] = g0[4]; o0[7] = g0[5];
    o1[0] = g0[6]; o1[1] = g0[7];
#pragma unroll
    for (int j = 0; j < 6; ++j) o1[2 + j] = g1[j];
    o2[0] = g1[6]; o2[1] = g1[7];
#pragma unroll
    for (int j = 2; j < 8; ++j) o2[j] = h0;
#pragma unroll
    for (int j = 0; j < 8; ++j) o3[j] = h0;
    _Float16* cr = colin + ((size_t)r * S_ALL + k) * 32;
    *reinterpret_cast<v8h*>(cr)      = o0;
    *reinterpret_cast<v8h*>(cr + 8)  = o1;
    *reinterpret_cast<v8h*>(cr + 16) = o2;
    *reinterpret_cast<v8h*>(cr + 24) = o3;
  }
}

// ================= K6: color MLP (18 -> 64 -> 3) via WMMA =================
__global__ __launch_bounds__(256) void k_color(
    const _Float16* __restrict__ colin,
    const _Float16* __restrict__ fragbuf,
    const float* __restrict__ bc1, const float* __restrict__ bc2,
    float* __restrict__ rgb) {              // [N*128*3]
  __shared__ __align__(16) _Float16 ldsH[8 * 16 * HID];
  const int lane = threadIdx.x & 31;
  const int wave = threadIdx.x >> 5;
  const int tile = blockIdx.x * 8 + wave;
  const int m    = lane & 15;
  const int sel  = lane >> 4;

  // A from packed inputs (K already padded to 32)
  v16h a = afrag_from_half(colin + (size_t)(tile * 16) * 32, 32, m, 0, lane);

  _Float16* myLds = ldsH + wave * 16 * HID;
#pragma unroll
  for (int nt = 0; nt < 4; ++nt) {
    v16h b = load_frag(fragbuf, 6 + nt, lane);
    v8f h = wmma_f32_16x16x32_f16(a, b, v8f_zero());
    const int n = nt * 16 + m;
    const float bb = bc1[n];
#pragma unroll
    for (int i = 0; i < 8; ++i) {
      float v = fmaxf(h[i] + bb, 0.0f);
      myLds[(sel * 8 + i) * HID + n] = (_Float16)v;
    }
  }
  __syncthreads();

  v8f o = v8f_zero();
#pragma unroll
  for (int c = 0; c < 2; ++c) {
    v16h a2 = afrag_from_half(myLds, HID, m, c * 32, lane);
    v16h b2 = load_frag(fragbuf, 10 + c, lane);
    o = wmma_f32_16x16x32_f16(a2, b2, o);
  }
  const int col = m;
  const float bo = (col < 3) ? bc2[col] : 0.0f;
#pragma unroll
  for (int i = 0; i < 8; ++i) {
    const int row = sel * 8 + i;
    const int smp = tile * 16 + row;
    const float v = o[i] + bo;
    if (col < 3) rgb[(size_t)smp * 3 + col] = 1.0f / (1.0f + expf(-v));
  }
}

// ================= K7: composite =================
__global__ void k_composite(const float* __restrict__ wts,
                            const float* __restrict__ rgb,
                            const float* __restrict__ bg,
                            float* __restrict__ out) {
  const int r = blockIdx.x * blockDim.x + threadIdx.x;
  if (r >= N_RAYS) return;
  float aR = 0.0f, aG = 0.0f, aB = 0.0f, wsum = 0.0f;
  for (int k = 0; k < S_ALL; ++k) {
    const float w = wts[(size_t)r * S_ALL + k];
    wsum += w;
    if (w > 1e-4f) {
      const size_t b = ((size_t)r * S_ALL + k) * 3;
      aR += w * rgb[b + 0];
      aG += w * rgb[b + 1];
      aB += w * rgb[b + 2];
    }
  }
  const float rem = 1.0f - wsum;
  out[r * 3 + 0] = aR + rem * bg[0];
  out[r * 3 + 1] = aG + rem * bg[1];
  out[r * 3 + 2] = aB + rem * bg[2];
}

// ================= launch =================
extern "C" void kernel_launch(void* const* d_in, const int* in_sizes, int n_in,
                              void* d_out, int out_size, void* d_ws, size_t ws_size,
                              hipStream_t stream) {
  const float* rays_o = (const float*)d_in[0];
  const float* rays_d = (const float*)d_in[1];
  const float* timep  = (const float*)d_in[2];
  const float* bg     = (const float*)d_in[3];
  const float* Wd1 = (const float*)d_in[4];
  const float* bd1 = (const float*)d_in[5];
  const float* Wd2 = (const float*)d_in[6];
  const float* bd2 = (const float*)d_in[7];
  const float* Wc1 = (const float*)d_in[8];
  const float* bc1 = (const float*)d_in[9];
  const float* Wc2 = (const float*)d_in[10];
  const float* bc2 = (const float*)d_in[11];

  // workspace carve-up (256B aligned)
  char* ws = (char*)d_ws;
  size_t off = 0;
  auto carve = [&](size_t bytes) {
    char* p = ws + off;
    off = (off + bytes + 255) & ~(size_t)255;
    return p;
  };
  _Float16* fragbuf = (_Float16*)carve((size_t)NFRAG * FRAG_HALVES * 2);
  float*    nears  = (float*)carve((size_t)N_RAYS * 4);
  float*    fars   = (float*)carve((size_t)N_RAYS * 4);
  float*    zc     = (float*)carve((size_t)N_RAYS * S_COARSE * 4);
  float*    zf     = (float*)carve((size_t)N_RAYS * S_FINE * 4);
  float*    sc     = (float*)carve((size_t)N_RAYS * S_COARSE * 4);
  float*    sf     = (float*)carve((size_t)N_RAYS * S_FINE * 4);
  _Float16* gc     = (_Float16*)carve((size_t)N_RAYS * S_COARSE * 16 * 2);
  _Float16* gf     = (_Float16*)carve((size_t)N_RAYS * S_FINE * 16 * 2);
  _Float16* colin  = (_Float16*)carve((size_t)N_RAYS * S_ALL * 32 * 2);
  float*    wts    = (float*)carve((size_t)N_RAYS * S_ALL * 4);
  float*    rgb    = (float*)carve((size_t)N_RAYS * S_ALL * 3 * 4);
  (void)ws_size;

  k_pack<<<NFRAG, 32, 0, stream>>>(Wd1, Wd2, Wc1, Wc2, fragbuf);

  k_prep<<<N_RAYS / 256, 256, 0, stream>>>(rays_o, rays_d, nears, fars, zc);

  const int densBlocks = (N_RAYS * S_COARSE / 16) / 8;  // 4096, exact
  k_density<<<densBlocks, 256, 0, stream>>>(rays_o, rays_d, timep, fragbuf,
                                            bd1, bd2, zc, sc, gc);

  k_importance<<<N_RAYS / 128, 128, 0, stream>>>(zc, sc, nears, fars, zf);

  k_density<<<densBlocks, 256, 0, stream>>>(rays_o, rays_d, timep, fragbuf,
                                            bd1, bd2, zf, sf, gf);

  k_merge<<<N_RAYS / 128, 128, 0, stream>>>(zc, zf, sc, sf, gc, gf, rays_d,
                                            nears, fars, colin, wts);

  const int colBlocks = (N_RAYS * S_ALL / 16) / 8;  // 8192, exact
  k_color<<<colBlocks, 256, 0, stream>>>(colin, fragbuf, bc1, bc2, rgb);

  k_composite<<<N_RAYS / 256, 256, 0, stream>>>(wts, rgb, bg, (float*)d_out);
}
